// GNNCommunicationLayer_8031588843645
// MI455X (gfx1250) — matrix operand order repaired and checked by reading, not compiled, not run on hardware
//
#include <hip/hip_runtime.h>
#include <cmath>

#define N_AGENTS 22
#define DD 128
#define TB 8
#define ROWS (TB * N_AGENTS)   // 176
#define MTILES (ROWS / 16)     // 11
#define XW 136                 // padded LDS row stride (bf16 elems) for x/h/msg
#define WEW 136                // padded stride for transposed W_enc / W_msg
#define WIW 264                // padded stride for transposed W_int (K=256)
#define SMEM_ELEMS (3 * ROWS * XW + 2 * DD * WEW + DD * WIW)
#define SMEM_BYTES (SMEM_ELEMS * 2)   // 280832 bytes < 320KB WGP LDS

typedef __attribute__((ext_vector_type(16))) __bf16       v16bf;
typedef __attribute__((ext_vector_type(8)))  float        v8f;
typedef __attribute__((ext_vector_type(4)))  unsigned int v4u;

union Frag { v4u u[2]; v16bf v; };

__device__ __forceinline__ float fast_tanh(float x) {
#if __has_builtin(__builtin_amdgcn_tanhf)
  return __builtin_amdgcn_tanhf(x);
#else
  float r;
  asm("v_tanh_f32 %0, %1\n\tv_nop" : "=v"(r) : "v"(x));
  return r;
#endif
}

__device__ __forceinline__ float bfu_lo(unsigned u) { return __uint_as_float(u << 16); }
__device__ __forceinline__ float bfu_hi(unsigned u) { return __uint_as_float(u & 0xffff0000u); }
__device__ __forceinline__ unsigned short bfbits(float f) {
  return __builtin_bit_cast(unsigned short, (__bf16)f);
}

// A fragment: 16x32 bf16, M rows striped per ISA layout.
// lane m (0..15): K = {k0..k0+7, k0+16..k0+23}; lane m+16: +8 on both halves.
__device__ __forceinline__ v16bf loadA(const __bf16* S, int r0, int k0, int lane) {
  const int m = lane & 15, hi = lane >> 4;
  const __bf16* p = S + (r0 + m) * XW + k0 + hi * 8;
  Frag f;
  f.u[0] = *(const v4u*)p;
  f.u[1] = *(const v4u*)(p + 16);
  return f.v;
}

__device__ __forceinline__ void loadA4(v16bf a[4], const __bf16* S, int r0, int lane) {
  #pragma unroll
  for (int kc = 0; kc < 4; ++kc) a[kc] = loadA(S, r0, kc * 32, lane);
}

// B fragment: 32x16 bf16 from TRANSPOSED weights Wt[n][k].
// lane n (0..15): K = {k0..k0+7, k0+16..k0+23}; lane n+16: +8 on both halves.
__device__ __forceinline__ v16bf loadB(const __bf16* Wt, int stride, int k0, int lane) {
  const int n = lane & 15, hi = lane >> 4;
  const __bf16* p = Wt + n * stride + k0 + hi * 8;
  Frag f;
  f.u[0] = *(const v4u*)p;
  f.u[1] = *(const v4u*)(p + 16);
  return f.v;
}

__global__ void __launch_bounds__(256, 1)
gnn_comm_kernel(const float* __restrict__ x,
                const float* __restrict__ Wenc, const float* __restrict__ benc,
                const float* __restrict__ Wmsg, const float* __restrict__ bmsg,
                const float* __restrict__ Wint, const float* __restrict__ bint,
                float* __restrict__ out) {
  extern __shared__ __bf16 smem[];
  __bf16* xs = smem;                 // [ROWS][XW]  x (bf16), later msg_out
  __bf16* hs = xs + ROWS * XW;       // [ROWS][XW]  h_local
  __bf16* ms = hs + ROWS * XW;       // [ROWS][XW]  messages
  __bf16* we = ms + ROWS * XW;       // [128][WEW]  W_enc^T
  __bf16* wm = we + DD * WEW;        // [128][WEW]  W_msg^T
  __bf16* wi = wm + DD * WEW;        // [128][WIW]  W_int^T (K=256)

  const int tid  = threadIdx.x;
  const int lane = tid & 31;
  const int w    = tid >> 5;        // wave id == N-tile (cols 16w..16w+15)
  const int col0 = w * 16;
  const int hi   = lane >> 4;
  const int nl   = lane & 15;
  const long rowBase = (long)blockIdx.x * ROWS;

  // ---- stage weights transposed as bf16 ----
  for (int j = 0; j < (DD * DD) / 256; ++j) {           // 64 iters
    int e = tid + 256 * j;
    int k = e >> 7, n = e & 127;                        // W[k][n] row-major
    we[n * WEW + k] = (__bf16)Wenc[e];
    wm[n * WEW + k] = (__bf16)Wmsg[e];
  }
  for (int j = 0; j < (2 * DD * DD) / 256; ++j) {       // 128 iters
    int e = tid + 256 * j;
    int k = e >> 7, n = e & 127;
    wi[n * WIW + k] = (__bf16)Wint[e];
  }
  // ---- stage x as bf16 (float4 coalesced loads, packed b32 LDS stores) ----
  {
    const float4* xg = (const float4*)(x + rowBase * DD);
    for (int j = 0; j < (ROWS * DD) / (256 * 4); ++j) { // 22 iters
      int e4 = tid + 256 * j;
      float4 v = xg[e4];
      int e = e4 * 4, r = e >> 7, c = e & 127;
      unsigned* p = (unsigned*)&xs[r * XW + c];
      p[0] = (unsigned)bfbits(v.x) | ((unsigned)bfbits(v.y) << 16);
      p[1] = (unsigned)bfbits(v.z) | ((unsigned)bfbits(v.w) << 16);
    }
  }
  __syncthreads();

  const float bbe = benc[col0 + nl];
  const float bbm = bmsg[col0 + nl];
  const float bbi = bint[col0 + nl];

  // ---- GEMM1: h = tanh(x @ W_enc + b_enc) ----
  {
    v16bf bfr[4];
    #pragma unroll
    for (int kc = 0; kc < 4; ++kc) bfr[kc] = loadB(we, WEW, kc * 32, lane);
    const v8f cIn = {bbe, bbe, bbe, bbe, bbe, bbe, bbe, bbe};
    v16bf cur[4], nxt[4];
    loadA4(cur, xs, 0, lane);
    #pragma unroll
    for (int mt = 0; mt < MTILES; ++mt) {
      if (mt + 1 < MTILES) loadA4(nxt, xs, (mt + 1) * 16, lane);
      v8f acc = __builtin_amdgcn_wmma_f32_16x16x32_bf16(false, cur[0], false, bfr[0],
                                                        (short)0, cIn, false, false);
      #pragma unroll
      for (int kc = 1; kc < 4; ++kc)
        acc = __builtin_amdgcn_wmma_f32_16x16x32_bf16(false, cur[kc], false, bfr[kc],
                                                      (short)0, acc, false, false);
      int rr = mt * 16 + hi * 8;
      #pragma unroll
      for (int r = 0; r < 8; ++r)
        hs[(rr + r) * XW + col0 + nl] = (__bf16)fast_tanh(acc[r]);
      #pragma unroll
      for (int kc = 0; kc < 4; ++kc) cur[kc] = nxt[kc];
    }
  }
  __syncthreads();

  // ---- messages = (sum_agents(h) - h) / (A-1), two dims per uint ----
  {
    const float inv = 1.0f / (float)(N_AGENTS - 1);
    #pragma unroll
    for (int j = 0; j < 2; ++j) {
      int p = tid + 256 * j;          // 8 batches x 64 dim-pairs = 512 groups
      int i = p >> 6, d = (p & 63) * 2;
      const unsigned* hr = (const unsigned*)&hs[(i * N_AGENTS) * XW + d];
      unsigned* mr = (unsigned*)&ms[(i * N_AGENTS) * XW + d];
      float s0 = 0.f, s1 = 0.f;
      #pragma unroll
      for (int a = 0; a < N_AGENTS; ++a) {
        unsigned u = hr[a * (XW / 2)];
        s0 += bfu_lo(u); s1 += bfu_hi(u);
      }
      #pragma unroll
      for (int a = 0; a < N_AGENTS; ++a) {
        unsigned u = hr[a * (XW / 2)];
        float m0 = (s0 - bfu_lo(u)) * inv;
        float m1 = (s1 - bfu_hi(u)) * inv;
        mr[a * (XW / 2)] = (unsigned)bfbits(m0) | ((unsigned)bfbits(m1) << 16);
      }
    }
  }
  __syncthreads();

  // ---- GEMM2: msg_out = tanh(messages @ W_msg + b_msg) -> overwrite xs ----
  {
    v16bf bfr[4];
    #pragma unroll
    for (int kc = 0; kc < 4; ++kc) bfr[kc] = loadB(wm, WEW, kc * 32, lane);
    const v8f cIn = {bbm, bbm, bbm, bbm, bbm, bbm, bbm, bbm};
    v16bf cur[4], nxt[4];
    loadA4(cur, ms, 0, lane);
    #pragma unroll
    for (int mt = 0; mt < MTILES; ++mt) {
      if (mt + 1 < MTILES) loadA4(nxt, ms, (mt + 1) * 16, lane);
      v8f acc = __builtin_amdgcn_wmma_f32_16x16x32_bf16(false, cur[0], false, bfr[0],
                                                        (short)0, cIn, false, false);
      #pragma unroll
      for (int kc = 1; kc < 4; ++kc)
        acc = __builtin_amdgcn_wmma_f32_16x16x32_bf16(false, cur[kc], false, bfr[kc],
                                                      (short)0, acc, false, false);
      int rr = mt * 16 + hi * 8;
      #pragma unroll
      for (int r = 0; r < 8; ++r)
        xs[(rr + r) * XW + col0 + nl] = (__bf16)fast_tanh(acc[r]);
      #pragma unroll
      for (int kc = 0; kc < 4; ++kc) cur[kc] = nxt[kc];
    }
  }
  __syncthreads();

  // ---- GEMM3: out = tanh([h | msg_out] @ W_int + b_int), K=256 ----
  {
    v16bf bfr[8];
    #pragma unroll
    for (int kc = 0; kc < 8; ++kc) bfr[kc] = loadB(wi, WIW, kc * 32, lane);
    const v8f cIn = {bbi, bbi, bbi, bbi, bbi, bbi, bbi, bbi};
    v16bf cur[8], nxt[8];
    loadA4(cur,     hs, 0, lane);
    loadA4(cur + 4, xs, 0, lane);
    #pragma unroll
    for (int mt = 0; mt < MTILES; ++mt) {
      if (mt + 1 < MTILES) {
        loadA4(nxt,     hs, (mt + 1) * 16, lane);
        loadA4(nxt + 4, xs, (mt + 1) * 16, lane);
      }
      v8f acc = __builtin_amdgcn_wmma_f32_16x16x32_bf16(false, cur[0], false, bfr[0],
                                                        (short)0, cIn, false, false);
      #pragma unroll
      for (int kc = 1; kc < 8; ++kc)
        acc = __builtin_amdgcn_wmma_f32_16x16x32_bf16(false, cur[kc], false, bfr[kc],
                                                      (short)0, acc, false, false);
      int rr = mt * 16 + hi * 8;
      float* og = out + (rowBase + rr) * DD + col0 + nl;
      #pragma unroll
      for (int r = 0; r < 8; ++r)
        og[(long)r * DD] = fast_tanh(acc[r]);
      #pragma unroll
      for (int kc = 0; kc < 8; ++kc) cur[kc] = nxt[kc];
    }
  }
}

extern "C" void kernel_launch(void* const* d_in, const int* in_sizes, int n_in,
                              void* d_out, int out_size, void* d_ws, size_t ws_size,
                              hipStream_t stream) {
  (void)n_in; (void)d_ws; (void)ws_size; (void)out_size;
  const float* x    = (const float*)d_in[0];
  const float* Wenc = (const float*)d_in[1];
  const float* benc = (const float*)d_in[2];
  const float* Wmsg = (const float*)d_in[3];
  const float* bmsg = (const float*)d_in[4];
  const float* Wint = (const float*)d_in[5];
  const float* bint = (const float*)d_in[6];
  float* out = (float*)d_out;

  int B = in_sizes[0] / (N_AGENTS * DD);   // 8192
  int blocks = B / TB;                      // 1024

  hipFuncSetAttribute(reinterpret_cast<const void*>(gnn_comm_kernel),
                      hipFuncAttributeMaxDynamicSharedMemorySize, SMEM_BYTES);
  gnn_comm_kernel<<<blocks, 256, SMEM_BYTES, stream>>>(x, Wenc, benc, Wmsg, bmsg,
                                                       Wint, bint, out);
}
